// Attention_67113158967542
// MI455X (gfx1250) — matrix operand order repaired and checked by reading, not compiled, not run on hardware
//
#include <hip/hip_runtime.h>
#include <stdint.h>

typedef _Float16 f16;
typedef __attribute__((ext_vector_type(16))) _Float16 v16h;
typedef __attribute__((ext_vector_type(8)))  float    v8f;

#define B_   2
#define S_   2048
#define H_   16
#define DK_  64
#define DM_  1024
#define NEGV (-65504.0f)

// ---------------------------------------------------------------------------
// WMMA + async helpers (gfx1250, wave32)
// ---------------------------------------------------------------------------
__device__ __forceinline__ v8f wmma_f16(v16h a, v16h b, v8f c) {
  return __builtin_amdgcn_wmma_f32_16x16x32_f16(false, a, false, b, (short)0, c,
                                                false, false);
}

__device__ __forceinline__ v8f zero8() {
  v8f z = {0.f, 0.f, 0.f, 0.f, 0.f, 0.f, 0.f, 0.f};
  return z;
}

// Async copy 16 bytes global -> LDS (GLOBAL_LOAD_ASYNC_TO_LDS_B128, ASYNCcnt).
__device__ __forceinline__ void async_b128(void* lds_ptr, const void* gptr) {
  uint32_t lds_off = (uint32_t)(uintptr_t)lds_ptr;  // addr[31:0] = LDS byte off
  unsigned long long ga = (unsigned long long)(uintptr_t)gptr;
  asm volatile("global_load_async_to_lds_b128 %0, %1, off"
               :
               : "v"(lds_off), "v"(ga)
               : "memory");
}
template <int N>
__device__ __forceinline__ void wait_async_le() {
  asm volatile("s_wait_asynccnt %0" ::"i"(N) : "memory");
}
__device__ __forceinline__ void wait_ds0() {
  asm volatile("s_wait_dscnt 0" ::: "memory");
}

// A-matrix 16x32 f16: lane m=L%16; lanes<16 K=0..15,
// lanes>=16 K=16..31.  base -> element [0][0], ld in elements.
__device__ __forceinline__ v16h load_a_f16(const f16* base, int ld, int lane) {
  int m = lane & 15, g = (lane >> 4) & 1;
  const f16* r = base + (size_t)m * ld + 16 * g;
  v16h a;
#pragma unroll
  for (int h = 0; h < 8; ++h) a[h] = r[h];
#pragma unroll
  for (int h = 0; h < 8; ++h) a[h + 8] = r[h + 8];
  return a;
}

// B-matrix 32x16 f16: lane col n=L%16; lanes<16 K=0..15, lanes>=16 K=16..31.
// base addresses B[k][n] at base[n*ld + k] (n-major, K-contiguous source).
__device__ __forceinline__ v16h load_b_f16(const f16* base, int ld, int lane) {
  int n = lane & 15, g = (lane >> 4) & 1;
  const f16* r = base + (size_t)n * ld + 16 * g;
  v16h b;
#pragma unroll
  for (int h = 0; h < 16; ++h) b[h] = r[h];
  return b;
}

__device__ __forceinline__ float red_max16(float v) {
#pragma unroll
  for (int off = 1; off < 16; off <<= 1) v = fmaxf(v, __shfl_xor(v, off, 32));
  return v;
}
__device__ __forceinline__ float red_sum16(float v) {
#pragma unroll
  for (int off = 1; off < 16; off <<= 1) v += __shfl_xor(v, off, 32);
  return v;
}

// ---------------------------------------------------------------------------
// One-shot f32 -> f16 conversion (memory bound, ~2us for 50MB at 23.3TB/s)
// ---------------------------------------------------------------------------
__global__ void __launch_bounds__(256)
cvt_f32_f16_kernel(const float* __restrict__ src, f16* __restrict__ dst, int n) {
  int i = (blockIdx.x * 256 + threadIdx.x) * 4;
  if (i + 3 < n) {
    float4 v = *(const float4*)(src + i);
    dst[i + 0] = (f16)v.x;
    dst[i + 1] = (f16)v.y;
    dst[i + 2] = (f16)v.z;
    dst[i + 3] = (f16)v.w;
  }
}

// ---------------------------------------------------------------------------
// All-f16 GEMM:  C[m,n] = sum_k A[m,k] * W[n,k]   (M=4096, N=1024, K=1024)
// Block = 4 waves sharing one 64-col N-group.  W tile (64x32 f16 = 4KB) is
// double-buffered in LDS via GLOBAL_LOAD_ASYNC_TO_LDS_B128: tile i+1 is
// issued before waiting (asynccnt<=2, in-order completion) so the copy
// overlaps compute.  Each wave owns TWO 16-row M-tiles -> every LDS B
// fragment feeds 2 WMMAs (8 wmma / K-step).
// mode 0: f16 out [B,H,S,64]; 1: f16 out [B,H,64,S]; 2: f32 out [4096,1024].
// ---------------------------------------------------------------------------
__device__ __forceinline__ void gemm_issue(const f16* __restrict__ W, f16* dst,
                                           int n0, int kk, int tid) {
#pragma unroll
  for (int u = 0; u < 2; ++u) {          // branch-free: exactly 2 chunks/thread
    int c = tid + u * 128;               // 256 x 16B chunks = 4KB tile
    int row = c >> 2, part = c & 3;      // 4 chunks per 64B row
    async_b128(dst + row * 32 + part * 8,
               W + (size_t)(n0 + row) * DM_ + kk + part * 8);
  }
}

__global__ void __launch_bounds__(128)
gemm_f16_kernel(const f16* __restrict__ A, const f16* __restrict__ W,
                f16* __restrict__ out16, float* __restrict__ out32, int mode) {
  __shared__ __align__(16) f16 sW[2][64 * 32];   // [buf][n_local 64][k 32]
  int tid = threadIdx.x;
  int lane = tid & 31, w = tid >> 5;
  int m0 = (blockIdx.x * 4 + w) * 32;   // two 16-row tiles per wave
  int n0 = blockIdx.y * 64;             // 16 N-groups
  int g = lane >> 4, nn = lane & 15;

  v8f acc0[4], acc1[4];
#pragma unroll
  for (int j = 0; j < 4; ++j) { acc0[j] = zero8(); acc1[j] = zero8(); }

  gemm_issue(W, sW[0], n0, 0, tid);     // prologue: stage tile 0

  for (int i = 0; i < 32; ++i) {
    int kk = i * 32;
    if (i + 1 < 32) {
      gemm_issue(W, sW[(i + 1) & 1], n0, kk + 32, tid);
      wait_async_le<2>();               // older tile's 2 chunks done (in-order)
    } else {
      wait_async_le<0>();
    }
    __syncthreads();

    const f16* sWb = sW[i & 1];
    v16h a0 = load_a_f16(A + (size_t)m0 * DM_ + kk, DM_, lane);
    v16h a1 = load_a_f16(A + (size_t)(m0 + 16) * DM_ + kk, DM_, lane);
#pragma unroll
    for (int j = 0; j < 4; ++j) {
      v16h bm = load_b_f16(sWb + (16 * j) * 32, 32, lane);
      acc0[j] = wmma_f16(a0, bm, acc0[j]);
      acc1[j] = wmma_f16(a1, bm, acc1[j]);
    }
    __syncthreads();                    // reads done before buffer reuse
  }

#pragma unroll
  for (int t = 0; t < 2; ++t) {
#pragma unroll
    for (int j = 0; j < 4; ++j) {
#pragma unroll
      for (int r = 0; r < 8; ++r) {
        float val = t ? acc1[j][r] : acc0[j][r];
        int m = m0 + 16 * t + r + 8 * g;   // global row (b*S + s)
        int n = n0 + 16 * j + nn;          // global col (h*64 + dk)
        if (mode == 2) {
          out32[(size_t)m * DM_ + n] = val;
        } else {
          int bb = m >> 11, s = m & (S_ - 1);
          int hh = n >> 6, dk = n & (DK_ - 1);
          if (mode == 1)
            out16[(((size_t)(bb * H_ + hh)) * DK_ + dk) * S_ + s] = (f16)val;
          else
            out16[(((size_t)(bb * H_ + hh)) * S_ + s) * DK_ + dk] = (f16)val;
        }
      }
    }
  }
}

// ---------------------------------------------------------------------------
// Flash attention: block = 4 waves sharing one (b,h); each wave owns a
// 16-query tile x all 2048 keys, online softmax.  K/V 32-key tiles are
// double-buffered in LDS via async copies (4x cross-wave reuse + overlap).
// Q,K: [B*H, S, 64] f16.  VT: [B*H, 64, S] f16.  mask: [B,S,S] int32.
// ---------------------------------------------------------------------------
__device__ __forceinline__ void attn_issue(const f16* __restrict__ Kg,
                                           const f16* __restrict__ VTg,
                                           f16* sKb, f16* sVb, int k0, int tid) {
#pragma unroll
  for (int u = 0; u < 2; ++u) {
    int c = tid + u * 128;               // 256 chunks each for K and VT
    int krow = c >> 3, kpart = c & 7;    // K tile: 32 rows x 128B
    async_b128(sKb + krow * 64 + kpart * 8,
               Kg + (size_t)(k0 + krow) * DK_ + kpart * 8);
    int vrow = c >> 2, vpart = c & 3;    // VT tile: 64 rows x 64B
    async_b128(sVb + vrow * 32 + vpart * 8,
               VTg + (size_t)vrow * S_ + k0 + vpart * 8);
  }
}

__global__ void __launch_bounds__(128)
attn_kernel(const f16* __restrict__ Q, const f16* __restrict__ K,
            const f16* __restrict__ VT, const int* __restrict__ mask,
            f16* __restrict__ Octx) {
  __shared__ __align__(16) f16 sK[2][32 * 64];     // [buf][k 32][d 64]
  __shared__ __align__(16) f16 sV[2][64 * 32];     // [buf][d 64][k 32]
  __shared__ __align__(16) f16 plds[4][16 * 32];   // per-wave P staging

  int tid = threadIdx.x;
  int lane = tid & 31, w = tid >> 5;
  int bh = blockIdx.y;                   // 0..31
  int b = bh >> 4, h = bh & 15;
  int q0 = (blockIdx.x * 4 + w) * 16;
  int g = lane >> 4, nn = lane & 15;
  f16* pl = plds[w];

  const f16* Kg  = K + (size_t)bh * S_ * DK_;
  const f16* VTg = VT + (size_t)bh * DK_ * S_;

  const f16* Qb = Q + ((size_t)bh * S_ + q0) * DK_;
  v16h a_lo = load_a_f16(Qb, DK_, lane);        // d = 0..31
  v16h a_hi = load_a_f16(Qb + 32, DK_, lane);   // d = 32..63

  v8f acc[4];
#pragma unroll
  for (int j = 0; j < 4; ++j) acc[j] = zero8();
  float mrow[8], lrow[8];
#pragma unroll
  for (int r = 0; r < 8; ++r) { mrow[r] = -3.0e38f; lrow[r] = 0.f; }

  const int* mb = mask + (size_t)b * S_ * S_;

  attn_issue(Kg, VTg, sK[0], sV[0], 0, tid);     // prologue: stage tile 0

  for (int i = 0; i < S_ / 32; ++i) {
    int k0 = i * 32;
    if (i + 1 < S_ / 32) {
      attn_issue(Kg, VTg, sK[(i + 1) & 1], sV[(i + 1) & 1], k0 + 32, tid);
      wait_async_le<4>();               // older tile's 4 chunks done (in-order)
    } else {
      wait_async_le<0>();
    }
    __syncthreads();

    const f16* sKb = sK[i & 1];
    const f16* sVb = sV[i & 1];

    // ---- scores for two 16-key sub-tiles (operands from LDS) ----
    v8f sc[2];
#pragma unroll
    for (int t = 0; t < 2; ++t) {
      v16h b0 = load_b_f16(sKb + (16 * t) * 64, DK_, lane);       // d 0..31
      v16h b1 = load_b_f16(sKb + (16 * t) * 64 + 32, DK_, lane);  // d 32..63
      v8f st = zero8();
      st = wmma_f16(a_lo, b0, st);
      st = wmma_f16(a_hi, b1, st);
#pragma unroll
      for (int r = 0; r < 8; ++r) {
        int q = q0 + r + 8 * g;
        int kcol = k0 + 16 * t + nn;
        int mv = mb[(size_t)q * S_ + kcol];
        float s = st[r] * 0.125f;                 // 1/sqrt(64)
        st[r] = (mv == 0) ? NEGV : s;
      }
      sc[t] = st;
    }

    // ---- online softmax (row lives in one VGPR across a 16-lane half) ----
#pragma unroll
    for (int r = 0; r < 8; ++r) {
      float tmax = red_max16(fmaxf(sc[0][r], sc[1][r]));
      float mnew = fmaxf(mrow[r], tmax);
      float fac = __expf(mrow[r] - mnew);
      float p0 = __expf(sc[0][r] - mnew);
      float p1 = __expf(sc[1][r] - mnew);
      float rs = red_sum16(p0 + p1);
      lrow[r] = lrow[r] * fac + rs;
      mrow[r] = mnew;
      sc[0][r] = p0;
      sc[1][r] = p1;
#pragma unroll
      for (int j = 0; j < 4; ++j) acc[j][r] *= fac;
    }

    // ---- stage P (16x32) through LDS: C-layout -> A-layout (wave-local) ----
#pragma unroll
    for (int r = 0; r < 8; ++r) {
      int row = r + 8 * g;
      pl[row * 32 + nn] = (f16)sc[0][r];
      pl[row * 32 + 16 + nn] = (f16)sc[1][r];
    }
    wait_ds0();
    v16h pa = load_a_f16(pl, 32, lane);

    // ---- O += P @ V ----
#pragma unroll
    for (int j = 0; j < 4; ++j) {
      v16h vb = load_b_f16(sVb + (16 * j) * 32, 32, lane);
      acc[j] = wmma_f16(pa, vb, acc[j]);
    }
    __syncthreads();   // all reads done before alternate-buffer reuse
  }

  // ---- normalize and store context (re-concat heads) ----
#pragma unroll
  for (int j = 0; j < 4; ++j) {
#pragma unroll
    for (int r = 0; r < 8; ++r) {
      int q = q0 + r + 8 * g;
      float o = acc[j][r] / lrow[r];
      Octx[((size_t)(b * S_ + q)) * DM_ + h * DK_ + 16 * j + nn] = (f16)o;
    }
  }
}

// ---------------------------------------------------------------------------
extern "C" void kernel_launch(void* const* d_in, const int* in_sizes, int n_in,
                              void* d_out, int out_size, void* d_ws,
                              size_t ws_size, hipStream_t stream) {
  const float* q   = (const float*)d_in[0];
  const float* k   = (const float*)d_in[1];
  const float* v   = (const float*)d_in[2];
  const int*  mask = (const int*)d_in[3];
  const float* w_q = (const float*)d_in[4];
  const float* w_k = (const float*)d_in[5];
  const float* w_v = (const float*)d_in[6];
  const float* w_o = (const float*)d_in[7];

  const size_t NACT = (size_t)B_ * S_ * DM_;   // 4,194,304
  const size_t NW   = (size_t)DM_ * DM_;       // 1,048,576
  f16* p = (f16*)d_ws;
  f16* Qws  = p; p += NACT;
  f16* Kws  = p; p += NACT;
  f16* VTws = p; p += NACT;
  f16* Octx = p; p += NACT;
  f16* qf   = p; p += NACT;
  f16* kf   = p; p += NACT;
  f16* vf   = p; p += NACT;
  f16* wq16 = p; p += NW;
  f16* wk16 = p; p += NW;
  f16* wv16 = p; p += NW;
  f16* wo16 = p; p += NW;   // total ~67 MB

  dim3 blk(128);
  // f32 -> f16 one-shot conversions
  int gact = (int)(NACT / 4 / 256), gw = (int)(NW / 4 / 256);
  cvt_f32_f16_kernel<<<gact, 256, 0, stream>>>(q, qf, (int)NACT);
  cvt_f32_f16_kernel<<<gact, 256, 0, stream>>>(k, kf, (int)NACT);
  cvt_f32_f16_kernel<<<gact, 256, 0, stream>>>(v, vf, (int)NACT);
  cvt_f32_f16_kernel<<<gw, 256, 0, stream>>>(w_q, wq16, (int)NW);
  cvt_f32_f16_kernel<<<gw, 256, 0, stream>>>(w_k, wk16, (int)NW);
  cvt_f32_f16_kernel<<<gw, 256, 0, stream>>>(w_v, wv16, (int)NW);
  cvt_f32_f16_kernel<<<gw, 256, 0, stream>>>(w_o, wo16, (int)NW);

  dim3 gemm_grid(32, 16);   // 32 blocks x 4 waves x 32 rows = 4096 M rows
  gemm_f16_kernel<<<gemm_grid, blk, 0, stream>>>(qf, wq16, Qws, nullptr, 0);
  gemm_f16_kernel<<<gemm_grid, blk, 0, stream>>>(kf, wk16, Kws, nullptr, 0);
  gemm_f16_kernel<<<gemm_grid, blk, 0, stream>>>(vf, wv16, VTws, nullptr, 1);
  attn_kernel<<<dim3(32, 32), blk, 0, stream>>>(Qws, Kws, VTws, mask, Octx);
  gemm_f16_kernel<<<gemm_grid, blk, 0, stream>>>(Octx, wo16, nullptr,
                                                 (float*)d_out, 2);
}
